// SharedNetWithPrototypes_50697793962000
// MI455X (gfx1250) — matrix operand order repaired and checked by reading, compiled, not run on hardware
//
#include <hip/hip_runtime.h>
#include <math.h>

// ---------------------------------------------------------------------------
// Types / helpers
// ---------------------------------------------------------------------------
typedef unsigned int   u32;
typedef unsigned short u16;
typedef __attribute__((ext_vector_type(16))) __bf16 v16bf;
typedef __attribute__((ext_vector_type(8)))  float  v8f;

union Frag { u32 u[8]; v16bf v; };

__device__ __forceinline__ float bf2f(u16 h) {
    u32 x = ((u32)h) << 16;
    return __builtin_bit_cast(float, x);
}
__device__ __forceinline__ u16 f2bf(float f) {
    u32 x = __builtin_bit_cast(u32, f);
    if ((x & 0x7F800000u) == 0x7F800000u) return (u16)(x >> 16); // inf/nan
    u32 r = (x + 0x7FFFu + ((x >> 16) & 1u)) >> 16;              // RNE
    return (u16)r;
}

#define GS_LOOP(i, total) \
    for (size_t i = (size_t)blockIdx.x * blockDim.x + threadIdx.x; i < (total); \
         i += (size_t)gridDim.x * blockDim.x)

static inline unsigned nblk(size_t total) {
    size_t b = (total + 255) / 256;
    return (unsigned)(b > 262144 ? 262144 : b);
}

// ---------------------------------------------------------------------------
// WMMA GEMM:  C[M,N] (f32, ld=N) = A[M,K] (bf16) * B[K,N] (bf16, ld=N)
// CONV==1: A gathered implicitly from NHWC bf16 activations (3x3 SAME conv,
//          k = (dy*3+dx)*Cin + c, Cin % 32 == 0). CONV==0: A row-major, lda=K.
// Block: 256 thr (8 waves). Tile 128x64, BK=32. Wave -> 32x32 (4 WMMA/step).
// ---------------------------------------------------------------------------
template <int CONV>
__global__ __launch_bounds__(256) void gemm_bf16(
    const u16* __restrict__ A, const u16* __restrict__ Bm, float* __restrict__ Cm,
    int M, int N, int K, int H, int W, int Cin)
{
    __shared__ __align__(16) u16 As[128][40];   // +8 pad: 16B-aligned rows, conflict-light
    __shared__ __align__(16) u16 Bs[64][40];    // stored transposed [n][k]

    const int tid  = threadIdx.x;
    const int m0   = blockIdx.y * 128;
    const int n0   = blockIdx.x * 64;
    const int lane = tid & 31,  wid = tid >> 5;
    const int wm   = (wid & 3) * 32, wn = (wid >> 2) * 32;
    const int half = lane >> 4, l16 = lane & 15;

    v8f acc[2][2];
#pragma unroll
    for (int s = 0; s < 2; ++s)
#pragma unroll
        for (int t = 0; t < 2; ++t) acc[s][t] = (v8f)0.0f;

    // A-tile loader coords: thread -> (row, 16-wide k half)
    const int arow = tid >> 1;
    const int kh   = (tid & 1) * 16;
    const int mA   = m0 + arow;
    int ab = 0, ay = 0, ax = 0;
    if (CONV) {
        const int hw = H * W;
        ab = mA / hw; int r = mA - ab * hw;
        ay = r / W;   ax = r - ay * W;
    }
    // B-tile loader coords (coalesced: consecutive lanes -> consecutive cols)
    const int bn2 = tid & 63;           // column within tile
    const int bk2 = tid >> 6;           // k base (0..3), step 4
    const int gnB = n0 + bn2;
    const int nnB = (gnB < N) ? gnB : (N - 1);   // clamped, always in-bounds
    const bool okB = gnB < N;

    for (int kk = 0; kk < K; kk += 32) {
        // ---- load A tile (16 bf16 per thread, zero-filled OOB) ----
        uint4 q0{}, q1{};
        const int kb = kk + kh;
        if (CONV) {
            const int p  = kb / Cin, c0 = kb - p * Cin;
            const int dy = p / 3 - 1, dx = p - (p / 3) * 3 - 1;
            const int iy = ay + dy,  ix = ax + dx;
            if (mA < M && iy >= 0 && iy < H && ix >= 0 && ix < W) {
                const uint4* g = (const uint4*)(A +
                    ((size_t)((ab * H + iy) * W + ix) * Cin + c0));
                q0 = g[0]; q1 = g[1];
            }
        } else {
            if (mA < M) {
                const uint4* g = (const uint4*)(A + ((size_t)mA * K + kb));
                q0 = g[0]; q1 = g[1];
                if (kk + 32 < K)
                    __builtin_prefetch(A + ((size_t)mA * K + kb + 32), 0, 3);
            }
        }
        *(uint4*)&As[arow][kh]     = q0;
        *(uint4*)&As[arow][kh + 8] = q1;

        // ---- load B tile transposed, branch-free guard, coalesced reads ----
#pragma unroll
        for (int kit = 0; kit < 8; ++kit) {
            const int k = bk2 + kit * 4;
            const u16 v = Bm[(size_t)(kk + k) * N + nnB];
            Bs[bn2][k] = okB ? v : (u16)0;
        }
        __syncthreads();

        // ---- build fragments per ISA VGPR layout, 4 WMMA ----
        Frag af[2], bf[2];
#pragma unroll
        for (int s = 0; s < 2; ++s) {
            const u16* ap = &As[wm + s * 16 + l16][0];
            const int  k0 = half * 8;
#pragma unroll
            for (int i = 0; i < 4; ++i) {
                af[s].u[i]     = *(const u32*)(ap + k0 + 2 * i);        // K 0..7 / 8..15
                af[s].u[4 + i] = *(const u32*)(ap + 16 + k0 + 2 * i);   // K 16..23 / 24..31
            }
            const u16* bp = &Bs[wn + s * 16 + l16][0];
            const int  kc = half * 8;
#pragma unroll
            for (int i = 0; i < 4; ++i) {
                bf[s].u[i]     = *(const u32*)(bp + kc + 2 * i);        // K 0..7 / 8..15
                bf[s].u[4 + i] = *(const u32*)(bp + 16 + kc + 2 * i);   // K 16..23 / 24..31
            }
        }
#pragma unroll
        for (int s = 0; s < 2; ++s)
#pragma unroll
            for (int t = 0; t < 2; ++t)
                acc[s][t] = __builtin_amdgcn_wmma_f32_16x16x32_bf16(
                    false, af[s].v, false, bf[t].v, (short)0, acc[s][t], false, false);
        __syncthreads();
    }

    // ---- store C (row = r + 8*half, col = l16) ----
#pragma unroll
    for (int s = 0; s < 2; ++s)
#pragma unroll
        for (int t = 0; t < 2; ++t)
#pragma unroll
            for (int r = 0; r < 8; ++r) {
                const int gm = m0 + wm + s * 16 + r + 8 * half;
                const int gn = n0 + wn + t * 16 + l16;
                if (gm < M && gn < N) Cm[(size_t)gm * N + gn] = acc[s][t][r];
            }
}

// ---------------------------------------------------------------------------
// Small kernels
// ---------------------------------------------------------------------------
__global__ void zero_f32(float* p, size_t n) { GS_LOOP(i, n) p[i] = 0.f; }

// sentinel im2col: x NCHW (B,2,72,72) f32 -> A bf16 [B*72*72, 32] (K 18 padded)
__global__ void im2col_sent(const float* __restrict__ x, u16* __restrict__ A,
                            int B, int H, int W) {
    const size_t total = (size_t)B * H * W * 32;
    GS_LOOP(i, total) {
        const int k = (int)(i & 31);
        const size_t m = i >> 5;
        const int b = (int)(m / (H * W));
        const int r = (int)(m - (size_t)b * H * W);
        const int y = r / W, xx = r - y * W;
        float v = 0.f;
        if (k < 18) {
            const int p = k >> 1, c = k & 1;
            const int iy = y + p / 3 - 1, ix = xx + p % 3 - 1;
            if (iy >= 0 && iy < H && ix >= 0 && ix < W)
                v = x[((size_t)(b * 2 + c) * H + iy) * W + ix];
        }
        A[i] = f2bf(v);
    }
}

// conv weight OIHW (O,C,3,3) f32 -> B bf16 [Kpad x O], k = p*C + c
__global__ void wconv_prep(const float* __restrict__ w, u16* __restrict__ Bm,
                           int O, int C, int Kpad) {
    const size_t total = (size_t)Kpad * O;
    GS_LOOP(i, total) {
        const int o = (int)(i % O);
        const int k = (int)(i / O);
        float v = 0.f;
        if (k < 9 * C) {
            const int p = k / C, c = k - p * C;
            v = w[((size_t)o * C + c) * 9 + p];
        }
        Bm[i] = f2bf(v);
    }
}

// FC weight (N,K) f32 -> B bf16 [K x N] (transpose)
__global__ void wfc_prep(const float* __restrict__ w, u16* __restrict__ Bm,
                         int Nn, int K) {
    const size_t total = (size_t)K * Nn;
    GS_LOOP(i, total) {
        const int n = (int)(i % Nn);
        const int k = (int)(i / Nn);
        Bm[i] = f2bf(w[(size_t)n * K + k]);
    }
}

// protos (300,96) -> [96 x 300] for S-GEMM
__global__ void protoS_prep(const float* __restrict__ pr, u16* __restrict__ Bm) {
    GS_LOOP(i, (size_t)96 * 300) {
        const int p = (int)(i % 300), c = (int)(i / 300);
        Bm[i] = f2bf(pr[(size_t)p * 96 + c]);
    }
}
// protos -> [320 x 96] (rows padded) for z-GEMM
__global__ void protoZ_prep(const float* __restrict__ pr, u16* __restrict__ Bm) {
    GS_LOOP(i, (size_t)320 * 96) {
        const int c = (int)(i % 96), p = (int)(i / 96);
        Bm[i] = (p < 300) ? f2bf(pr[(size_t)p * 96 + c]) : (u16)0;
    }
}

// per-column sum/sumsq partials (coalesced 32-col tiles) + atomics
__global__ void colstats_partial(const float* __restrict__ X, int M, int N,
                                 int rowsPerBlk, float* __restrict__ sum,
                                 float* __restrict__ sq) {
    __shared__ float s1[8][32], s2[8][32];
    const int c  = blockIdx.x * 32 + threadIdx.x;
    const int r0 = blockIdx.y * rowsPerBlk;
    const int r1 = min(M, r0 + rowsPerBlk);
    float a = 0.f, b = 0.f;
    if (c < N)
        for (int m = r0 + threadIdx.y; m < r1; m += 8) {
            const float v = X[(size_t)m * N + c];
            a += v; b += v * v;
        }
    s1[threadIdx.y][threadIdx.x] = a;
    s2[threadIdx.y][threadIdx.x] = b;
    __syncthreads();
    if (threadIdx.y == 0 && c < N) {
        float A = 0.f, B = 0.f;
#pragma unroll
        for (int j = 0; j < 8; ++j) { A += s1[j][threadIdx.x]; B += s2[j][threadIdx.x]; }
        atomicAdd(&sum[c], A);
        atomicAdd(&sq[c], B);
    }
}
__global__ void colstats_final(const float* sum, const float* sq,
                               float* mean, float* var, int M, int N) {
    GS_LOOP(i, (size_t)N) {
        const float m = sum[i] / (float)M;
        mean[i] = m;
        var[i]  = sq[i] / (float)M - m * m;
    }
}

// (x-mean)*rsqrt(var+eps), relu; optional f32 and/or bf16 outputs
__global__ void bn_relu(const float* __restrict__ X, const float* __restrict__ mean,
                        const float* __restrict__ var, size_t M, int N,
                        float* __restrict__ outf, u16* __restrict__ outh) {
    const size_t total = M * N;
    GS_LOOP(i, total) {
        const int c = (int)(i % N);
        float v = (X[i] - mean[c]) * rsqrtf(var[c] + 1e-5f);
        v = fmaxf(v, 0.f);
        if (outf) outf[i] = v;
        if (outh) outh[i] = f2bf(v);
    }
}

// 3x3 stride-2 VALID maxpool, NHWC f32 -> NHWC bf16
__global__ void maxpool32(const float* __restrict__ in, u16* __restrict__ out,
                          int B, int H, int W, int C, int Ho, int Wo) {
    const size_t total = (size_t)B * Ho * Wo * C;
    GS_LOOP(i, total) {
        const int c = (int)(i % C);
        size_t t = i / C;
        const int ox = (int)(t % Wo); t /= Wo;
        const int oy = (int)(t % Ho);
        const int b  = (int)(t / Ho);
        float m = -3.4e38f;
#pragma unroll
        for (int ky = 0; ky < 3; ++ky)
#pragma unroll
            for (int kx = 0; kx < 3; ++kx) {
                const int iy = oy * 2 + ky, ix = ox * 2 + kx;
                m = fmaxf(m, in[((size_t)(b * H + iy) * W + ix) * C + c]);
            }
        out[i] = f2bf(m);
    }
}

// L2-normalize rows (wave per row), bf16 -> bf16
__global__ void rownorm(const u16* __restrict__ in, u16* __restrict__ out,
                        int M, int C) {
    const int gw   = (int)((blockIdx.x * blockDim.x + threadIdx.x) >> 5);
    const int lane = threadIdx.x & 31;
    if (gw >= M) return;
    float s = 0.f;
    for (int j = lane; j < C; j += 32) {
        const float v = bf2f(in[(size_t)gw * C + j]);
        s += v * v;
    }
#pragma unroll
    for (int o = 16; o >= 1; o >>= 1) s += __shfl_xor(s, o, 32);
    const float sc = 1.f / fmaxf(sqrtf(s), 1e-12f);
    for (int j = lane; j < C; j += 32)
        out[(size_t)gw * C + j] = f2bf(bf2f(in[(size_t)gw * C + j]) * sc);
}

__global__ void sinkK(const float* __restrict__ S, float* __restrict__ K, size_t n) {
    GS_LOOP(i, n) K[i] = __expf(-(1.f - S[i]) * 10.f);   // REG = 0.1
}
__global__ void fill_ones(float* p, size_t n) { GS_LOOP(i, n) p[i] = 1.f; }

// v[b,j] = (1/300) / sum_i K[b,i,j]*u[b,i]
__global__ void sink_v(const float* __restrict__ K, const float* __restrict__ u,
                       float* __restrict__ v) {
    __shared__ float us[1225];
    const int b = blockIdx.y;
    for (int i = threadIdx.x; i < 1225; i += 256) us[i] = u[(size_t)b * 1225 + i];
    __syncthreads();
    const int j = blockIdx.x * 256 + threadIdx.x;
    if (j < 300) {
        const float* Kb = K + (size_t)b * 1225 * 300 + j;
        float acc = 0.f;
        for (int i = 0; i < 1225; ++i) acc += Kb[(size_t)i * 300] * us[i];
        v[(size_t)b * 300 + j] = (1.f / 300.f) / acc;
    }
}
// u[b,i] = (1/1225) / sum_j K[b,i,j]*v[b,j]   (wave per row)
__global__ void sink_u(const float* __restrict__ K, const float* __restrict__ v,
                       float* __restrict__ u) {
    __shared__ float vs[300];
    const int b = blockIdx.y;
    for (int j = threadIdx.x; j < 300; j += 256) vs[j] = v[(size_t)b * 300 + j];
    __syncthreads();
    const int wid  = threadIdx.x >> 5, lane = threadIdx.x & 31;
    const int i    = blockIdx.x * 8 + wid;
    if (i >= 1225) return;
    const float* Kr = K + ((size_t)b * 1225 + i) * 300;
    float acc = 0.f;
    for (int j = lane; j < 300; j += 32) acc += Kr[j] * vs[j];
#pragma unroll
    for (int o = 16; o >= 1; o >>= 1) acc += __shfl_xor(acc, o, 32);
    if (lane == 0) u[(size_t)b * 1225 + i] = (1.f / 1225.f) / acc;
}

// gamma (scaled by batch=128) -> bf16 A [M x 320], zero-padded cols
__global__ void gamma_pack(const float* __restrict__ K, const float* __restrict__ u,
                           const float* __restrict__ v, u16* __restrict__ A, int M) {
    const size_t total = (size_t)M * 320;
    GS_LOOP(i, total) {
        const int j = (int)(i % 320);
        const size_t m = i / 320;
        float val = 0.f;
        if (j < 300) {
            const int b = (int)(m / 1225);
            val = 128.f * u[m] * K[m * 300 + j] * v[(size_t)b * 300 + j];
        }
        A[i] = f2bf(val);
    }
}

__global__ void f32_to_bf16(const float* __restrict__ in, u16* __restrict__ out, size_t n) {
    GS_LOOP(i, n) out[i] = f2bf(in[i]);
}

// NHWC (B,8,8,192) bf16 -> flat (B, C*H*W) bf16 in NCHW flatten order
__global__ void flatten_pack(const u16* __restrict__ in, u16* __restrict__ out) {
    GS_LOOP(i, (size_t)128 * 12288) {
        const int b = (int)(i / 12288), k = (int)(i % 12288);
        const int c = k / 64, rem = k % 64, y = rem / 8, x = rem % 8;
        out[i] = in[((size_t)(b * 8 + y) * 8 + x) * 192 + c];
    }
}

// out = (C7 + b7) / ||.||2 per row  (128x128)
__global__ void bias_l2norm(const float* __restrict__ C7, const float* __restrict__ b7,
                            float* __restrict__ out) {
    __shared__ float red[128];
    const int r = blockIdx.x, t = threadIdx.x;
    const float v = C7[(size_t)r * 128 + t] + b7[t];
    red[t] = v * v;
    __syncthreads();
    for (int s = 64; s >= 1; s >>= 1) {
        if (t < s) red[t] += red[t + s];
        __syncthreads();
    }
    out[(size_t)r * 128 + t] = v * rsqrtf(red[0]);
}

// ---------------------------------------------------------------------------
// Orchestration
// ---------------------------------------------------------------------------
extern "C" void kernel_launch(void* const* d_in, const int* in_sizes, int n_in,
                              void* d_out, int out_size, void* d_ws, size_t ws_size,
                              hipStream_t stream) {
    (void)in_sizes; (void)n_in; (void)out_size; (void)ws_size;
    const float* x     = (const float*)d_in[0];
    const float* sentw = (const float*)d_in[1];
    const float* prot  = (const float*)d_in[2];
    const float* w1    = (const float*)d_in[3];
    const float* w2    = (const float*)d_in[4];
    const float* w3    = (const float*)d_in[5];
    const float* w4    = (const float*)d_in[6];
    const float* w5    = (const float*)d_in[7];
    const float* w6    = (const float*)d_in[8];
    const float* w7    = (const float*)d_in[9];
    const float* b7    = (const float*)d_in[10];
    float* outp = (float*)d_out;

    // bump allocator over d_ws
    size_t off = 0;
    auto alloc = [&](size_t bytes) -> char* {
        char* p = (char*)d_ws + off;
        off = (off + bytes + 255) & ~(size_t)255;
        return p;
    };
    const int Bq = 128;
    const int Msent = Bq * 72 * 72;      // 663552
    const int M1    = Bq * 35 * 35;      // 156800
    const int M2    = Bq * 17 * 17;      // 36992

    // weights (bf16), persistent for the whole call
    u16* Bsent = (u16*)alloc((size_t)32 * 96 * 2);
    u16* B1    = (u16*)alloc((size_t)864 * 192 * 2);
    u16* B2    = (u16*)alloc((size_t)1728 * 384 * 2);
    u16* B3    = (u16*)alloc((size_t)3456 * 384 * 2);
    u16* B4    = (u16*)alloc((size_t)3456 * 192 * 2);
    u16* BpS   = (u16*)alloc((size_t)96 * 300 * 2);
    u16* BpZ   = (u16*)alloc((size_t)320 * 96 * 2);
    u16* B5    = (u16*)alloc((size_t)12288 * 4096 * 2);
    u16* B6    = (u16*)alloc((size_t)4096 * 4096 * 2);
    u16* B7    = (u16*)alloc((size_t)4096 * 128 * 2);

    // reusable scratch arenas (lifetimes interleaved, see schedule below)
    char* X0 = alloc((size_t)Msent * 32 * 2);        //  42.5 MB
    char* X1 = alloc((size_t)Msent * 96 * 4);        // 254.8 MB
    char* X2 = alloc((size_t)M1 * 96 * 2);           //  30.1 MB
    char* X3 = alloc((size_t)M1 * 300 * 4);          // 188.2 MB
    float* stats = (float*)alloc((size_t)4 * 4096 * 4);
    float* ubuf  = (float*)alloc((size_t)Bq * 1225 * 4);
    float* vbuf  = (float*)alloc((size_t)Bq * 300 * 4);
    float* ssum = stats, *ssq = stats + 4096, *smean = stats + 8192, *svar = stats + 12288;

    auto gemm_rm = [&](const u16* A, const u16* Bm, float* C, int M, int N, int K) {
        dim3 g((N + 63) / 64, (M + 127) / 128);
        gemm_bf16<0><<<g, 256, 0, stream>>>(A, Bm, C, M, N, K, 0, 0, 0);
    };
    auto gemm_cv = [&](const u16* A, const u16* Bm, float* C, int H, int W, int Cin, int N) {
        const int M = Bq * H * W, K = 9 * Cin;
        dim3 g((N + 63) / 64, (M + 127) / 128);
        gemm_bf16<1><<<g, 256, 0, stream>>>(A, Bm, C, M, N, K, H, W, Cin);
    };
    auto stats_of = [&](const float* X, int M, int N) {
        zero_f32<<<nblk(8192), 256, 0, stream>>>(ssum, 8192);
        dim3 blk(32, 8), grd((N + 31) / 32, (M + 8191) / 8192);
        colstats_partial<<<grd, blk, 0, stream>>>(X, M, N, 8192, ssum, ssq);
        colstats_final<<<nblk(N), 256, 0, stream>>>(ssum, ssq, smean, svar, M, N);
    };
    auto bn = [&](const float* X, int M, int N, float* of, u16* oh) {
        bn_relu<<<nblk((size_t)M * N), 256, 0, stream>>>(X, smean, svar, (size_t)M, N, of, oh);
    };

    // ---- weight prep (deterministic, every call) ----
    wconv_prep<<<nblk(32 * 96), 256, 0, stream>>>(sentw, Bsent, 96, 2, 32);
    wconv_prep<<<nblk(864 * 192), 256, 0, stream>>>(w1, B1, 192, 96, 864);
    wconv_prep<<<nblk(1728 * 384), 256, 0, stream>>>(w2, B2, 384, 192, 1728);
    wconv_prep<<<nblk(3456 * 384), 256, 0, stream>>>(w3, B3, 384, 384, 3456);
    wconv_prep<<<nblk(3456 * 192), 256, 0, stream>>>(w4, B4, 192, 384, 3456);
    protoS_prep<<<nblk(96 * 300), 256, 0, stream>>>(prot, BpS);
    protoZ_prep<<<nblk(320 * 96), 256, 0, stream>>>(prot, BpZ);
    wfc_prep<<<nblk((size_t)12288 * 4096), 256, 0, stream>>>(w5, B5, 4096, 12288);
    wfc_prep<<<nblk((size_t)4096 * 4096), 256, 0, stream>>>(w6, B6, 4096, 4096);
    wfc_prep<<<nblk((size_t)4096 * 128), 256, 0, stream>>>(w7, B7, 128, 4096);

    // ---- sentinel: conv(2->96) + BN + ReLU + pool ----
    im2col_sent<<<nblk((size_t)Msent * 32), 256, 0, stream>>>(x, (u16*)X0, Bq, 72, 72);
    gemm_rm((u16*)X0, Bsent, (float*)X1, Msent, 96, 32);
    stats_of((float*)X1, Msent, 96);
    bn((float*)X1, Msent, 96, (float*)X1, nullptr);
    maxpool32<<<nblk((size_t)M1 * 96), 256, 0, stream>>>((float*)X1, (u16*)X2, Bq, 72, 72, 96, 35, 35);

    // ---- prototypes: S = norm(feat) @ protos^T ----
    rownorm<<<(M1 + 7) / 8, 256, 0, stream>>>((u16*)X2, (u16*)X0, M1, 96);
    gemm_rm((u16*)X0, BpS, (float*)X1, M1, 300, 96);          // S  (156800 x 300)

    // ---- Sinkhorn (10 iters) ----
    sinkK<<<nblk((size_t)M1 * 300), 256, 0, stream>>>((float*)X1, (float*)X3, (size_t)M1 * 300);
    fill_ones<<<nblk((size_t)Bq * 1225), 256, 0, stream>>>(ubuf, (size_t)Bq * 1225);
    for (int it = 0; it < 10; ++it) {
        sink_v<<<dim3(2, Bq), 256, 0, stream>>>((float*)X3, ubuf, vbuf);
        sink_u<<<dim3((1225 + 7) / 8, Bq), 256, 0, stream>>>((float*)X3, vbuf, ubuf);
    }
    gamma_pack<<<nblk((size_t)M1 * 320), 256, 0, stream>>>((float*)X3, ubuf, vbuf, (u16*)X1, M1);
    gemm_rm((u16*)X1, BpZ, (float*)X3, M1, 96, 320);          // z  (156800 x 96)
    f32_to_bf16<<<nblk((size_t)M1 * 96), 256, 0, stream>>>((float*)X3, (u16*)X0, (size_t)M1 * 96);

    // ---- conv1 (96->192) + BN + ReLU + pool ----
    gemm_cv((u16*)X0, B1, (float*)X1, 35, 35, 96, 192);
    stats_of((float*)X1, M1, 192);
    bn((float*)X1, M1, 192, (float*)X1, nullptr);
    maxpool32<<<nblk((size_t)M2 * 192), 256, 0, stream>>>((float*)X1, (u16*)X2, Bq, 35, 35, 192, 17, 17);

    // ---- conv2 (192->384) + BN + ReLU ----
    gemm_cv((u16*)X2, B2, (float*)X3, 17, 17, 192, 384);
    stats_of((float*)X3, M2, 384);
    bn((float*)X3, M2, 384, nullptr, (u16*)X0);

    // ---- conv3 (384->384) + BN + ReLU ----
    gemm_cv((u16*)X0, B3, (float*)X1, 17, 17, 384, 384);
    stats_of((float*)X1, M2, 384);
    bn((float*)X1, M2, 384, nullptr, (u16*)X2);

    // ---- conv4 (384->192) + BN + ReLU + pool ----
    gemm_cv((u16*)X2, B4, (float*)X3, 17, 17, 384, 192);
    stats_of((float*)X3, M2, 192);
    bn((float*)X3, M2, 192, (float*)X3, nullptr);
    maxpool32<<<nblk((size_t)Bq * 64 * 192), 256, 0, stream>>>((float*)X3, (u16*)X0, Bq, 17, 17, 192, 8, 8);

    // ---- flatten (NCHW order) + FC stack ----
    flatten_pack<<<nblk((size_t)Bq * 12288), 256, 0, stream>>>((u16*)X0, (u16*)X1);
    gemm_rm((u16*)X1, B5, (float*)X3, Bq, 4096, 12288);
    stats_of((float*)X3, Bq, 4096);
    bn((float*)X3, Bq, 4096, nullptr, (u16*)X0);
    gemm_rm((u16*)X0, B6, (float*)X1, Bq, 4096, 4096);
    stats_of((float*)X1, Bq, 4096);
    bn((float*)X1, Bq, 4096, nullptr, (u16*)X2);
    gemm_rm((u16*)X2, B7, (float*)X3, Bq, 128, 4096);
    bias_l2norm<<<128, 128, 0, stream>>>((float*)X3, b7, outp);
}
